// SimpleBot_59648505807127
// MI455X (gfx1250) — compile-verified, matrix operand order checked by reading, not run on hardware
//
#include <hip/hip_runtime.h>
#include <hip/hip_bf16.h>

// ---------------------------------------------------------------------------
// Types for WMMA fragments (wave32, v_wmma_f32_16x16x32_f16)
// ---------------------------------------------------------------------------
typedef __attribute__((ext_vector_type(16))) _Float16 v16h;
typedef __attribute__((ext_vector_type(8)))  _Float16 v8h;
typedef __attribute__((ext_vector_type(4)))  _Float16 v4h;
typedef __attribute__((ext_vector_type(8)))  float    v8f;
typedef __attribute__((ext_vector_type(4)))  int      v4i;

__device__ __forceinline__ v16h ld_frag(const _Float16* p) {
  // ISA 7.12.2 16-bit A/B layout: lane holds K = kb..kb+7 and kb+16..kb+23
  union { v16h v; v8h h[2]; } u;
  u.h[0] = *(const v8h*)(p);
  u.h[1] = *(const v8h*)(p + 16);
  return u.v;
}

__device__ __forceinline__ v8f wmma_f16(v16h a, v16h b, v8f c) {
  return __builtin_amdgcn_wmma_f32_16x16x32_f16(
      /*neg_a=*/false, a, /*neg_b=*/false, b,
      /*c_mod=*/(short)0, c, /*reuse_a=*/false, /*reuse_b=*/false);
}

// Async global->LDS staging (gfx1250 ASYNCcnt path) with safe fallback.
// Probe: builtin exists, arity 4; param 0 is a generic `int4*` (from hipcc
// diagnostic). Pass LDS side as addrspace(3) int4* (implicitly converts to
// generic if the param is generic, exact match if it is AS(3)).
#if defined(__has_builtin)
#if __has_builtin(__builtin_amdgcn_global_load_async_to_lds_b128) && \
    __has_builtin(__builtin_amdgcn_s_wait_asynccnt)
#define USE_ASYNC_LDS 1
#endif
#endif

__device__ __forceinline__ void cp16_lds(_Float16* lds, const _Float16* g) {
#ifdef USE_ASYNC_LDS
  __builtin_amdgcn_global_load_async_to_lds_b128(
      (v4i*)g, (__attribute__((address_space(3))) v4i*)lds, 0, 0);
#else
  *(v8h*)lds = *(const v8h*)g;
#endif
}
__device__ __forceinline__ void async_fence() {
#ifdef USE_ASYNC_LDS
  __builtin_amdgcn_s_wait_asynccnt(0);
#endif
}

#define H 512
#define H2 1024
#define B 128
#define TD 20
#define TE 20
#define V 40000
// LDS row stride: 520 halves = 260 dwords == 4 (mod 64 banks) -> the 16 rows'
// b128 reads tile all 64 banks with no aliasing (528 would alias rows r, r+8).
#define LSTR 520

// ---------------------------------------------------------------------------
// Weight convert: f32 (K x N) row-major -> f16 (N x K) row-major (transposed)
// ---------------------------------------------------------------------------
__global__ void __launch_bounds__(256)
tr_cvt(const float* __restrict__ src, _Float16* __restrict__ dst, int K, int N) {
  __shared__ float t[32][33];
  const int n0 = blockIdx.x * 32, k0 = blockIdx.y * 32;
  #pragma unroll
  for (int j = 0; j < 32; j += 8) {
    int k = k0 + threadIdx.y + j, n = n0 + threadIdx.x;
    if (k < K && n < N) t[threadIdx.y + j][threadIdx.x] = src[(size_t)k * N + n];
  }
  __syncthreads();
  #pragma unroll
  for (int j = 0; j < 32; j += 8) {
    int n = n0 + threadIdx.y + j, k = k0 + threadIdx.x;
    if (n < N && k < K) dst[(size_t)n * K + k] = (_Float16)t[threadIdx.x][threadIdx.y + j];
  }
}

// ---------------------------------------------------------------------------
// Embedding gather: one block per (t,b) row, 128 threads x float4 -> 4 halves
// ---------------------------------------------------------------------------
__global__ void __launch_bounds__(128)
gather_k(const int* __restrict__ toks, const float* __restrict__ emb,
         _Float16* __restrict__ out) {
  const int r = blockIdx.x, j = threadIdx.x * 4;
  const int tok = toks[r];
  float4 v = *(const float4*)(emb + (size_t)tok * H + j);
  v4h o = {(_Float16)v.x, (_Float16)v.y, (_Float16)v.z, (_Float16)v.w};
  *(v4h*)(out + (size_t)r * H + j) = o;
}

__global__ void __launch_bounds__(256)
zero_k(_Float16* p, int n8) {
  int i = blockIdx.x * 256 + threadIdx.x;
  if (i < n8) { v8h z = {}; *(v8h*)(p + (size_t)i * 8) = z; }
}

// ---------------------------------------------------------------------------
// GRU gate: RU = sigmoid([x,h] @ Wg + bg)   (B x 2H), K = 2H split as x|h
// grid (8,8) x 256 threads: blockIdx.y = M-tile, wave handles one 16x16 N-tile
// ---------------------------------------------------------------------------
__global__ void __launch_bounds__(256)
gru_gate_k(const _Float16* __restrict__ X, const _Float16* __restrict__ Hin,
           const _Float16* __restrict__ Wt /* (2H x 2H) transposed: N x K */,
           const float* __restrict__ bg, float* __restrict__ RU) {
  __shared__ __align__(16) _Float16 Xs[16][LSTR];
  __shared__ __align__(16) _Float16 Hs[16][LSTR];
  const int tid = threadIdx.x;
  const int mbase = blockIdx.y * 16;
  for (int it = tid; it < 16 * 64; it += 256) {   // stage A operands in LDS
    int r = it >> 6, c = (it & 63) << 3;
    cp16_lds(&Xs[r][c], X + (size_t)(mbase + r) * H + c);
    cp16_lds(&Hs[r][c], Hin + (size_t)(mbase + r) * H + c);
  }
  async_fence();
  __syncthreads();
  const int wave = tid >> 5, lane = tid & 31;
  const int l15 = lane & 15, kb = (lane >> 4) << 3;
  const int n0 = (blockIdx.x * 8 + wave) * 16;
  const _Float16* bp = Wt + (size_t)(n0 + l15) * H2 + kb;
  v8f acc = {};
  #pragma unroll
  for (int kc = 0; kc < 16; ++kc)       // x segment: K = 0..511
    acc = wmma_f16(ld_frag(&Xs[l15][kc * 32 + kb]), ld_frag(bp + kc * 32), acc);
  #pragma unroll
  for (int kc = 0; kc < 16; ++kc)       // h segment: K = 512..1023
    acc = wmma_f16(ld_frag(&Hs[l15][kc * 32 + kb]), ld_frag(bp + H + kc * 32), acc);
  const int n = n0 + l15;
  const float bias = bg[n];
  const int rb = (lane >> 4) << 3;      // C layout: lanes>=16 hold rows M+8
  #pragma unroll
  for (int i = 0; i < 8; ++i) {
    int m = mbase + rb + i;
    float vv = acc[i] + bias;
    RU[(size_t)m * H2 + n] = 1.0f / (1.0f + __expf(-vv));
  }
}

// rh = r * h  (f16), r = RU[:, 0:H]
__global__ void __launch_bounds__(256)
gru_prep_k(const float* __restrict__ RU, const _Float16* __restrict__ Hin,
           _Float16* __restrict__ RH) {
  int i = blockIdx.x * 256 + threadIdx.x;         // over B*H = 65536
  int b = i >> 9, j = i & 511;
  RH[i] = (_Float16)(RU[(size_t)b * H2 + j] * (float)Hin[i]);
}

// ---------------------------------------------------------------------------
// GRU candidate + combine: c = tanh([x, r*h] @ Wc + bc); h' = u*h + (1-u)*c
// grid (4,8) x 256 threads
// ---------------------------------------------------------------------------
__global__ void __launch_bounds__(256)
gru_cand_k(const _Float16* __restrict__ X, const _Float16* __restrict__ RH,
           const _Float16* __restrict__ Hin,
           const _Float16* __restrict__ Wt /* (H x 2H) transposed: N x K */,
           const float* __restrict__ bc, const float* __restrict__ RU,
           _Float16* __restrict__ Hout, _Float16* __restrict__ Yout, int write_y) {
  __shared__ __align__(16) _Float16 Xs[16][LSTR];
  __shared__ __align__(16) _Float16 Rs[16][LSTR];
  const int tid = threadIdx.x;
  const int mbase = blockIdx.y * 16;
  for (int it = tid; it < 16 * 64; it += 256) {
    int r = it >> 6, c = (it & 63) << 3;
    cp16_lds(&Xs[r][c], X + (size_t)(mbase + r) * H + c);
    cp16_lds(&Rs[r][c], RH + (size_t)(mbase + r) * H + c);
  }
  async_fence();
  __syncthreads();
  const int wave = tid >> 5, lane = tid & 31;
  const int l15 = lane & 15, kb = (lane >> 4) << 3;
  const int n0 = (blockIdx.x * 8 + wave) * 16;
  const _Float16* bp = Wt + (size_t)(n0 + l15) * H2 + kb;
  v8f acc = {};
  #pragma unroll
  for (int kc = 0; kc < 16; ++kc)
    acc = wmma_f16(ld_frag(&Xs[l15][kc * 32 + kb]), ld_frag(bp + kc * 32), acc);
  #pragma unroll
  for (int kc = 0; kc < 16; ++kc)
    acc = wmma_f16(ld_frag(&Rs[l15][kc * 32 + kb]), ld_frag(bp + H + kc * 32), acc);
  const int n = n0 + l15;
  const float bias = bc[n];
  const int rb = (lane >> 4) << 3;
  #pragma unroll
  for (int i = 0; i < 8; ++i) {
    int m = mbase + rb + i;
    float c = tanhf(acc[i] + bias);
    float u = RU[(size_t)m * H2 + H + n];
    float h0 = (float)Hin[(size_t)m * H + n];
    float hn = u * h0 + (1.0f - u) * c;
    Hout[(size_t)m * H + n] = (_Float16)hn;
    if (write_y) Yout[(size_t)m * H + n] = (_Float16)hn;
  }
}

// ---------------------------------------------------------------------------
// Fused logits + log-softmax + CE (flash-softmax): never materializes logits.
// 80 blocks x 256 threads; block owns 32 rows of dec_outs (two 16-row M-tiles
// sharing every B fragment -> 2 WMMAs per B load, halving L2 traffic), streams
// all 2500 16-wide N-tiles of W_out^T with online max / sum-exp tracking.
// ---------------------------------------------------------------------------
__global__ void __launch_bounds__(256)
ce_k(const _Float16* __restrict__ Y /* (TD*B, H) */,
     const _Float16* __restrict__ Wt /* (V, H) */,
     const float* __restrict__ bo, const int* __restrict__ dect /* (TD+1,B) */,
     float* __restrict__ ce) {
  __shared__ __align__(16) _Float16 As[32][LSTR];
  __shared__ float rm[8][32], rs[8][32], rt[8][32];
  __shared__ int tgts[32];
  const int tid = threadIdx.x;
  const int mbase = blockIdx.x * 32;
  for (int it = tid; it < 32 * 64; it += 256) {
    int r = it >> 6, c = (it & 63) << 3;
    cp16_lds(&As[r][c], Y + (size_t)(mbase + r) * H + c);
  }
  if (tid < 32) tgts[tid] = dect[B + mbase + tid];  // targets = dec_tokens[1:]
  async_fence();
  __syncthreads();
  const int wave = tid >> 5, lane = tid & 31;
  const int l15 = lane & 15, kb = (lane >> 4) << 3, rb = (lane >> 4) << 3;
  float mx[16], sm[16], tl[16];
  #pragma unroll
  for (int i = 0; i < 16; ++i) { mx[i] = -3.0e38f; sm[i] = 0.0f; tl[i] = -3.0e38f; }
  for (int nt = wave; nt < V / 16; nt += 8) {
    const int n0 = nt * 16;
    const _Float16* bp = Wt + (size_t)(n0 + l15) * H + kb;
    __builtin_prefetch(bp + (size_t)128 * H, 0, 0);   // this wave's next B tile
    v8f acc0 = {}, acc1 = {};
    #pragma unroll
    for (int kc = 0; kc < 16; ++kc) {
      v16h bfrag = ld_frag(bp + kc * 32);             // shared by both M-tiles
      acc0 = wmma_f16(ld_frag(&As[l15][kc * 32 + kb]), bfrag, acc0);
      acc1 = wmma_f16(ld_frag(&As[16 + l15][kc * 32 + kb]), bfrag, acc1);
    }
    const int n = n0 + l15;
    const float bias = bo[n];
    #pragma unroll
    for (int i = 0; i < 8; ++i) {
      float v0 = acc0[i] + bias;
      float v1 = acc1[i] + bias;
      if (n == tgts[rb + i])      tl[i] = v0;         // capture target logits
      if (n == tgts[16 + rb + i]) tl[8 + i] = v1;
      float nm0 = fmaxf(mx[i], v0);                   // online softmax, tile 0
      sm[i] = sm[i] * __expf(mx[i] - nm0) + __expf(v0 - nm0);
      mx[i] = nm0;
      float nm1 = fmaxf(mx[8 + i], v1);               // online softmax, tile 1
      sm[8 + i] = sm[8 + i] * __expf(mx[8 + i] - nm1) + __expf(v1 - nm1);
      mx[8 + i] = nm1;
    }
  }
  // reduce across the 16 lanes sharing a row set (xor masks stay in halves)
  #pragma unroll
  for (int off = 1; off < 16; off <<= 1) {
    #pragma unroll
    for (int i = 0; i < 16; ++i) {
      float om = __shfl_xor(mx[i], off, 32);
      float os = __shfl_xor(sm[i], off, 32);
      float ot = __shfl_xor(tl[i], off, 32);
      float nm = fmaxf(mx[i], om);
      sm[i] = sm[i] * __expf(mx[i] - nm) + os * __expf(om - nm);
      mx[i] = nm;
      tl[i] = fmaxf(tl[i], ot);
    }
  }
  if (l15 == 0) {
    #pragma unroll
    for (int i = 0; i < 8; ++i) {
      rm[wave][rb + i] = mx[i];      rs[wave][rb + i] = sm[i];      rt[wave][rb + i] = tl[i];
      rm[wave][16 + rb + i] = mx[8 + i]; rs[wave][16 + rb + i] = sm[8 + i]; rt[wave][16 + rb + i] = tl[8 + i];
    }
  }
  __syncthreads();
  if (tid < 32) {       // combine 8 waves, emit CE for this row
    float m = -3.0e38f, s = 0.0f, t = -3.0e38f;
    #pragma unroll
    for (int w = 0; w < 8; ++w) {
      float nm = fmaxf(m, rm[w][tid]);
      s = s * __expf(m - nm) + rs[w][tid] * __expf(rm[w][tid] - nm);
      m = nm;
      t = fmaxf(t, rt[w][tid]);
    }
    ce[mbase + tid] = -(t - m - __logf(s));
  }
}

// Final weighted reduction to scalar loss
__global__ void __launch_bounds__(128)
loss_k(const float* __restrict__ ce, const float* __restrict__ w,
       float* __restrict__ out) {
  __shared__ float red[128];
  const int b = threadIdx.x;
  float s = 0.0f, ws = 0.0f;
  for (int t = 0; t < TD; ++t) {
    int idx = t * B + b;
    s += ce[idx] * w[idx];
    ws += w[idx];
  }
  red[b] = s / (ws + 1e-12f);
  __syncthreads();
  for (int off = 64; off > 0; off >>= 1) {
    if (b < off) red[b] += red[b + off];
    __syncthreads();
  }
  if (b == 0) out[0] = red[0] / (float)B;
}

// ---------------------------------------------------------------------------
// Host orchestration
// ---------------------------------------------------------------------------
extern "C" void kernel_launch(void* const* d_in, const int* in_sizes, int n_in,
                              void* d_out, int out_size, void* d_ws, size_t ws_size,
                              hipStream_t stream) {
  const int*   enc_tokens = (const int*)d_in[0];
  const int*   dec_tokens = (const int*)d_in[1];
  const float* tw         = (const float*)d_in[2];
  const float* emb        = (const float*)d_in[3];
  const float* Wg_enc     = (const float*)d_in[4];
  const float* bg_enc     = (const float*)d_in[5];
  const float* Wc_enc     = (const float*)d_in[6];
  const float* bc_enc     = (const float*)d_in[7];
  const float* Wg_dec     = (const float*)d_in[8];
  const float* bg_dec     = (const float*)d_in[9];
  const float* Wc_dec     = (const float*)d_in[10];
  const float* bc_dec     = (const float*)d_in[11];
  const float* W_out      = (const float*)d_in[12];
  const float* b_out      = (const float*)d_in[13];

  char* ws = (char*)d_ws;
  size_t off = 0;
  auto alloc = [&](size_t bytes) -> char* {
    char* p = ws + off;
    off += (bytes + 255) & ~(size_t)255;
    return p;
  };
  _Float16* WgeT = (_Float16*)alloc((size_t)H2 * H2 * 2);   // 2 MB
  _Float16* WceT = (_Float16*)alloc((size_t)H  * H2 * 2);   // 1 MB
  _Float16* WgdT = (_Float16*)alloc((size_t)H2 * H2 * 2);   // 2 MB
  _Float16* WcdT = (_Float16*)alloc((size_t)H  * H2 * 2);   // 1 MB
  _Float16* WoT  = (_Float16*)alloc((size_t)V  * H  * 2);   // 41 MB
  _Float16* encE = (_Float16*)alloc((size_t)TE * B * H * 2);
  _Float16* decE = (_Float16*)alloc((size_t)TD * B * H * 2);
  _Float16* hA   = (_Float16*)alloc((size_t)B * H * 2);
  _Float16* hB   = (_Float16*)alloc((size_t)B * H * 2);
  _Float16* RH   = (_Float16*)alloc((size_t)B * H * 2);
  float*    RU   = (float*)   alloc((size_t)B * H2 * 4);
  _Float16* decO = (_Float16*)alloc((size_t)TD * B * H * 2);
  float*    ceb  = (float*)   alloc((size_t)TD * B * 4);

  dim3 tb(32, 8);
  tr_cvt<<<dim3(H2 / 32, H2 / 32), tb, 0, stream>>>(Wg_enc, WgeT, H2, H2);
  tr_cvt<<<dim3(H  / 32, H2 / 32), tb, 0, stream>>>(Wc_enc, WceT, H2, H);
  tr_cvt<<<dim3(H2 / 32, H2 / 32), tb, 0, stream>>>(Wg_dec, WgdT, H2, H2);
  tr_cvt<<<dim3(H  / 32, H2 / 32), tb, 0, stream>>>(Wc_dec, WcdT, H2, H);
  tr_cvt<<<dim3(V  / 32, H  / 32), tb, 0, stream>>>(W_out,  WoT,  H,  V);

  gather_k<<<TE * B, 128, 0, stream>>>(enc_tokens, emb, encE);
  gather_k<<<TD * B, 128, 0, stream>>>(dec_tokens, emb, decE);   // rows 0..TD-1
  zero_k<<<(B * H / 8 + 255) / 256, 256, 0, stream>>>(hA, B * H / 8);

  _Float16* hcur = hA;
  _Float16* hnxt = hB;
  for (int t = 0; t < TE; ++t) {     // encoder scan
    gru_gate_k<<<dim3(8, 8), 256, 0, stream>>>(encE + (size_t)t * B * H, hcur, WgeT, bg_enc, RU);
    gru_prep_k<<<B * H / 256, 256, 0, stream>>>(RU, hcur, RH);
    gru_cand_k<<<dim3(4, 8), 256, 0, stream>>>(encE + (size_t)t * B * H, RH, hcur, WceT,
                                               bc_enc, RU, hnxt, hnxt, 0);
    _Float16* tmp = hcur; hcur = hnxt; hnxt = tmp;
  }
  for (int t = 0; t < TD; ++t) {     // decoder scan (hcur == enc_state)
    gru_gate_k<<<dim3(8, 8), 256, 0, stream>>>(decE + (size_t)t * B * H, hcur, WgdT, bg_dec, RU);
    gru_prep_k<<<B * H / 256, 256, 0, stream>>>(RU, hcur, RH);
    gru_cand_k<<<dim3(4, 8), 256, 0, stream>>>(decE + (size_t)t * B * H, RH, hcur, WcdT,
                                               bc_dec, RU, hnxt, decO + (size_t)t * B * H, 1);
    _Float16* tmp = hcur; hcur = hnxt; hnxt = tmp;
  }

  ce_k<<<TD * B / 32, 256, 0, stream>>>(decO, WoT, b_out, dec_tokens, ceb);
  loss_k<<<1, 128, 0, stream>>>(ceb, tw, (float*)d_out);
}
